// LSTM_66675072303478
// MI455X (gfx1250) — matrix operand order repaired and checked by reading, not compiled, not run on hardware
//
#include <hip/hip_runtime.h>

#define BATCH 512
#define SEQ   128
#define EIN   7
#define HID   1024
#define GATE  4096
#define OUTN  672        // 96 * 7
#define KC0   1056       // [h0(1024) | x(7) | pad(25)]
#define P0    1064       // LDS pitch (bf16 elems): 532 dwords -> conflict-free b128 reads
#define KC1   2048       // [h0_new(1024) | h1(1024)]
#define P1    2056       // 1028 dwords -> conflict-free b128 reads

typedef __attribute__((ext_vector_type(16))) __bf16 v16bf;
typedef __attribute__((ext_vector_type(8)))  __bf16 v8bf;
typedef __attribute__((ext_vector_type(8)))  float  v8f;

__device__ __forceinline__ v8f v8f_splat(float v) {
  v8f r;
#pragma unroll
  for (int i = 0; i < 8; ++i) r[i] = v;
  return r;
}

__device__ __forceinline__ float fast_sig(float v) {
  return __builtin_amdgcn_rcpf(1.0f + __expf(-v));
}
__device__ __forceinline__ float fast_tanh(float v) {
  const float e = __expf(2.0f * v);
  return (e - 1.0f) * __builtin_amdgcn_rcpf(e + 1.0f);
}

// A-fragment: lane m (lane&15) holds row m,
// K contiguous {k0+off .. +16}, off = (lane>>4)*16 (matches B).
__device__ __forceinline__ v16bf load_a_frag(const __bf16* p) {
  const v8bf lo = *(const v8bf*)p;
  const v8bf hi = *(const v8bf*)(p + 8);
  return __builtin_shufflevector(lo, hi, 0, 1, 2, 3, 4, 5, 6, 7,
                                 8, 9, 10, 11, 12, 13, 14, 15);
}

// One GEMM pass: 4 gates x 2 sub-tiles, K = nkk*32.
// wB0 covers gate rows g=0,1 ; wB2 covers g=2,3. Tile selection is a
// compile-time element offset that folds into the VMEM instruction offset
// (max byte offset (1024+48)*2048*2 = 4.39MB < 2^23).
template <int KC, int SOFF>
__device__ __forceinline__ void mma_block(v8f (&acc)[4][2],
                                          const __bf16* __restrict__ aP,
                                          const __bf16* __restrict__ wB0,
                                          const __bf16* __restrict__ wB2,
                                          int nkk) {
  for (int kk = 0; kk < nkk; ++kk) {
    const int k0 = kk * 32;
    const v16bf a = load_a_frag(aP + k0);
#pragma unroll
    for (int g = 0; g < 4; ++g)
#pragma unroll
      for (int s2 = 0; s2 < 2; ++s2) {
        const __bf16* bp =
            (g < 2) ? wB0 + (size_t)(g * HID + SOFF + s2 * 16) * KC
                    : wB2 + (size_t)((g - 2) * HID + SOFF + s2 * 16) * KC;
        const v16bf b = *(const v16bf*)(bp + k0);
        acc[g][s2] = __builtin_amdgcn_wmma_f32_16x16x32_bf16(
            false, a, false, b, (short)0, acc[g][s2], false, false);
      }
  }
}

// Combine gates for this pass's 2 sub-tiles; update c-state, emit h.
__device__ __forceinline__ void lstm_act(const v8f (&acc)[4][2], v8f* cst,
                                         float (*hv)[8]) {
#pragma unroll
  for (int s2 = 0; s2 < 2; ++s2)
#pragma unroll
    for (int i = 0; i < 8; ++i) {
      const float ig = fast_sig(acc[0][s2][i]);
      const float fg = fast_sig(acc[1][s2][i]);
      const float gg = fast_tanh(acc[2][s2][i]);
      const float og = fast_sig(acc[3][s2][i]);
      const float c  = fg * cst[s2][i] + ig * gg;
      cst[s2][i] = c;
      hv[s2][i]  = og * fast_tanh(c);
    }
}

// ---------------- weight pre-pass: f32 -> bf16, concatenated layouts ----------
__global__ void cvt_cat0(const float* __restrict__ Wh0,
                         const float* __restrict__ Wi0,
                         __bf16* __restrict__ out) {
  const int idx = blockIdx.x * blockDim.x + threadIdx.x;
  if (idx >= GATE * KC0) return;
  const int n = idx / KC0;
  const int k = idx - n * KC0;
  float v = 0.0f;
  if (k < HID)            v = Wh0[(size_t)n * HID + k];
  else if (k < HID + EIN) v = Wi0[(size_t)n * EIN + (k - HID)];
  out[idx] = (__bf16)v;
}

__global__ void cvt_cat1(const float* __restrict__ Wi1,
                         const float* __restrict__ Wh1,
                         __bf16* __restrict__ out) {
  const int idx = blockIdx.x * blockDim.x + threadIdx.x;
  if (idx >= GATE * KC1) return;
  const int n = idx / KC1;
  const int k = idx - n * KC1;
  const float v = (k < HID) ? Wi1[(size_t)n * HID + k]
                            : Wh1[(size_t)n * HID + (k - HID)];
  out[idx] = (__bf16)v;
}

__global__ void cvt_fc(const float* __restrict__ Wfc, __bf16* __restrict__ out) {
  const int idx = blockIdx.x * blockDim.x + threadIdx.x;
  if (idx >= OUTN * HID) return;
  out[idx] = (__bf16)Wfc[idx];
}

// ---------------- persistent fused 2-layer LSTM + FC head --------------------
__launch_bounds__(512, 1)
__global__ void lstm2_wmma(const float* __restrict__ x,
                           const float* __restrict__ b0,
                           const float* __restrict__ b1,
                           const float* __restrict__ bfc,
                           const __bf16* __restrict__ Wc0,
                           const __bf16* __restrict__ Wc1,
                           const __bf16* __restrict__ Wf,
                           float* __restrict__ out) {
  __shared__ __align__(32) __bf16 A0[16 * P0];  // [h0 | x_t | 0-pad]
  __shared__ __align__(32) __bf16 A1[16 * P1];  // [h0_new | h1]

  const int tid   = threadIdx.x;
  const int lane  = tid & 31;
  const int wv    = tid >> 5;        // wave 0..15, owns hidden cols [wv*64, +64)
  const int nIdx  = lane & 15;
  const int half  = lane >> 4;
  const int cRow  = half * 8;        // C/D layout: lanes 16-31 hold rows 8..15
  const int aOff  = half * 16;       // A layout K sub-offset
  const int bOff  = half * 16;       // B layout K sub-offset
  const int bbase = blockIdx.x << 4; // 16 batch rows per workgroup

  for (int i = tid; i < 16 * P0; i += 512) A0[i] = (__bf16)0.0f;
  for (int i = tid; i < 16 * P1; i += 512) A1[i] = (__bf16)0.0f;

  // two base pointers per layer; tile row selection is a constant offset
  const __bf16* w0lo = Wc0 + (size_t)(wv * 64 + nIdx) * KC0 + bOff;
  const __bf16* w0hi = w0lo + (size_t)2 * HID * KC0;
  const __bf16* w1lo = Wc1 + (size_t)(wv * 64 + nIdx) * KC1 + bOff;
  const __bf16* w1hi = w1lo + (size_t)2 * HID * KC1;
  const float* bq0 = b0 + wv * 64 + nIdx;   // + g*1024 + s*16 per tile
  const float* bq1 = b1 + wv * 64 + nIdx;

  v8f c0[4], c1[4];
#pragma unroll
  for (int s = 0; s < 4; ++s) { c0[s] = v8f_splat(0.0f); c1[s] = v8f_splat(0.0f); }

  const __bf16* aP0 = A0 + nIdx * P0 + aOff;
  const __bf16* aP1 = A1 + nIdx * P1 + aOff;

  __syncthreads();

  for (int t = 0; t < SEQ; ++t) {
    // stage x_t (16 rows x 7 feats) into the K-tail of the layer-0 A buffer
    if (tid < 16 * EIN) {
      const int r = tid / EIN;
      const int e = tid - r * EIN;
      A0[r * P0 + HID + e] = (__bf16)x[((size_t)(bbase + r) * SEQ + t) * EIN + e];
    }
    __syncthreads();  // x_t + h1(prev) visible

    // ---- layer 0: gates0 = [h0 | x_t] @ [Wh0|Wi0]^T + b0 ---- (2 passes)
    float h0v[4][8];
    v8f acc[4][2];
    {
#pragma unroll
      for (int g = 0; g < 4; ++g)
#pragma unroll
        for (int s2 = 0; s2 < 2; ++s2)
          acc[g][s2] = v8f_splat(bq0[g * HID + 0 + s2 * 16]);
      mma_block<KC0, 0>(acc, aP0, w0lo, w0hi, KC0 / 32);
      lstm_act(acc, &c0[0], &h0v[0]);

#pragma unroll
      for (int g = 0; g < 4; ++g)
#pragma unroll
        for (int s2 = 0; s2 < 2; ++s2)
          acc[g][s2] = v8f_splat(bq0[g * HID + 32 + s2 * 16]);
      mma_block<KC0, 32>(acc, aP0, w0lo, w0hi, KC0 / 32);
      lstm_act(acc, &c0[2], &h0v[2]);
    }

    __syncthreads();  // all waves done reading A0 & A1-front
#pragma unroll
    for (int s = 0; s < 4; ++s)
#pragma unroll
      for (int i = 0; i < 8; ++i) {
        const int r   = cRow + i;
        const int cix = wv * 64 + s * 16 + nIdx;
        const __bf16 hv = (__bf16)h0v[s][i];
        A0[r * P0 + cix] = hv;  // next step's recurrent input
        A1[r * P1 + cix] = hv;  // layer-1 input this step
      }
    __syncthreads();  // h0_new visible

    // ---- layer 1: gates1 = [h0_new | h1] @ [Wi1|Wh1]^T + b1 ---- (2 passes)
    float h1v[4][8];
    {
#pragma unroll
      for (int g = 0; g < 4; ++g)
#pragma unroll
        for (int s2 = 0; s2 < 2; ++s2)
          acc[g][s2] = v8f_splat(bq1[g * HID + 0 + s2 * 16]);
      mma_block<KC1, 0>(acc, aP1, w1lo, w1hi, KC1 / 32);
      lstm_act(acc, &c1[0], &h1v[0]);

#pragma unroll
      for (int g = 0; g < 4; ++g)
#pragma unroll
        for (int s2 = 0; s2 < 2; ++s2)
          acc[g][s2] = v8f_splat(bq1[g * HID + 32 + s2 * 16]);
      mma_block<KC1, 32>(acc, aP1, w1lo, w1hi, KC1 / 32);
      lstm_act(acc, &c1[2], &h1v[2]);
    }

    __syncthreads();  // all waves done reading A1-back
#pragma unroll
    for (int s = 0; s < 4; ++s)
#pragma unroll
      for (int i = 0; i < 8; ++i) {
        const int r   = cRow + i;
        const int cix = wv * 64 + s * 16 + nIdx;
        A1[r * P1 + HID + cix] = (__bf16)h1v[s][i];
      }
    // next iteration's first barrier publishes h1_new
  }

  __syncthreads();  // final h1 visible in A1-back

  // ---- FC head: pred = h1_last @ Wfc^T + bfc  (672 = 42 n-tiles) ----
  for (int nt = wv; nt < OUTN / 16; nt += 16) {
    v8f acc = v8f_splat(bfc[nt * 16 + nIdx]);
    const __bf16* wp = Wf + (size_t)(nt * 16 + nIdx) * HID + bOff;
    for (int kk = 0; kk < HID / 32; ++kk) {
      const int k0 = kk * 32;
      const v16bf a = load_a_frag(aP1 + HID + k0);
      const v16bf b = *(const v16bf*)(wp + k0);
      acc = __builtin_amdgcn_wmma_f32_16x16x32_bf16(
          false, a, false, b, (short)0, acc, false, false);
    }
#pragma unroll
    for (int i = 0; i < 8; ++i)
      out[(size_t)(bbase + cRow + i) * OUTN + nt * 16 + nIdx] = acc[i];
  }
}

// ---------------------------------------------------------------------------
extern "C" void kernel_launch(void* const* d_in, const int* in_sizes, int n_in,
                              void* d_out, int out_size, void* d_ws, size_t ws_size,
                              hipStream_t stream) {
  (void)in_sizes; (void)n_in; (void)out_size; (void)ws_size;
  const float* x   = (const float*)d_in[0];
  const float* Wi0 = (const float*)d_in[1];
  const float* Wh0 = (const float*)d_in[2];
  const float* b0  = (const float*)d_in[3];
  const float* Wi1 = (const float*)d_in[4];
  const float* Wh1 = (const float*)d_in[5];
  const float* b1  = (const float*)d_in[6];
  const float* Wfc = (const float*)d_in[7];
  const float* bfc = (const float*)d_in[8];
  float* out = (float*)d_out;

  char* ws = (char*)d_ws;
  const size_t szc0 = (size_t)GATE * KC0 * sizeof(__bf16);  // ~8.65 MB
  const size_t szc1 = (size_t)GATE * KC1 * sizeof(__bf16);  // ~16.8 MB
  __bf16* Wc0 = (__bf16*)ws;
  __bf16* Wc1 = (__bf16*)(ws + szc0);
  __bf16* Wf  = (__bf16*)(ws + szc0 + szc1);                // ~1.4 MB

  {
    const int n = GATE * KC0;
    cvt_cat0<<<(n + 255) / 256, 256, 0, stream>>>(Wh0, Wi0, Wc0);
  }
  {
    const int n = GATE * KC1;
    cvt_cat1<<<(n + 255) / 256, 256, 0, stream>>>(Wi1, Wh1, Wc1);
  }
  {
    const int n = OUTN * HID;
    cvt_fc<<<(n + 255) / 256, 256, 0, stream>>>(Wfc, Wf);
  }
  lstm2_wmma<<<BATCH / 16, 512, 0, stream>>>(x, b0, b1, bfc, Wc0, Wc1, Wf, out);
}
